// DTIPredictorV4_6373731467767
// MI455X (gfx1250) — matrix operand order repaired and checked by reading, not compiled, not run on hardware
//
#include <hip/hip_runtime.h>
#include <cstdint>
#include <cstddef>

// ---------------------------------------------------------------------------
// MI455X (gfx1250) implementation of the DTI GNN forward pass.
// Dense GEMMs: v_wmma_f32_16x16x32_f16, register-tiled NT N-tiles per wave
// (A fragments loaded once per K-step, 7x less A traffic) with the weight
// slab staged in LDS per block (ds_load_b128-fed B fragments).
// bias + LeakyReLU + BatchNorm fused in the epilogue.
// Feature dims padded: 200 -> 224, 400 -> 448, 600 -> 672 (zero pads are
// invariant through the pipeline). Dim-1 projections use wave32 dot-reductions.
// ---------------------------------------------------------------------------

typedef _Float16 h16;
typedef __attribute__((ext_vector_type(16))) _Float16 v16h;
typedef __attribute__((ext_vector_type(8)))  _Float16 v8h;
typedef __attribute__((ext_vector_type(8)))  float    v8f;

#define GPAD  224          // G=200 padded (mult of 32 for K, 16 for N)
#define G3PAD 672          // 3 GRU gate blocks of 224
#define G2PAD 448          // 2 blocks of 224
#define BN_SCALE 0.9999950000374997f  // 1/sqrt(1 + 1e-5)

namespace {
constexpr int cN1 = 25600, cE1 = 102400;   // ligand nodes / edges
constexpr int cN2 = 51200, cE2 = 204800;   // pocket nodes / edges
constexpr int cN3 = 76800, cE3 = 256000;   // interaction graph
constexpr int cB  = 256;                   // batch (num graphs)
}

__device__ __forceinline__ float d_lrelu(float v) { return v > 0.f ? v : 0.01f * v; }
__device__ __forceinline__ float d_sig(float v)   { return 1.f / (1.f + expf(-v)); }

__device__ __forceinline__ void atomicMaxF(float* addr, float v) {
  if (v >= 0.f) atomicMax((int*)addr, __float_as_int(v));
  else          atomicMin((unsigned int*)addr, __float_as_uint(v));
}

// ---------------------------------------------------------------------------
// WMMA GEMM: C[M,NP] = act(A[M,KP] @ W[NP,KP]^T + bias) * bn_g + bn_b
// Block = 256 threads = 8 waves = 8 M-tiles; each wave computes NT N-tiles.
// The block's B slab (NT*16 rows x KP) is staged in LDS once; the K-loop
// loads one A fragment (2 x b128 global) and feeds NT WMMAs from LDS.
// A fragment per ISA 7.12.2 (16-bit A 16x32); B per-lane uses the same k order as A.
// ---------------------------------------------------------------------------
template <int NT>
__global__ __launch_bounds__(256)
void dti_gemm_t(const h16* __restrict__ A, const h16* __restrict__ W,
                const float* __restrict__ bias,
                h16* __restrict__ Ch, float* __restrict__ Cf,
                int M, int KP, int NP, int act,
                const float* __restrict__ bng, const float* __restrict__ bnb) {
  extern __shared__ h16 ldsB[];
  const int tid = threadIdx.x;
  const int nb = blockIdx.x * NT * 16;       // first weight row for this block
  // cooperative stage of B slab [NT*16, KP] into LDS (v8h granules)
  const int chunksPerRow = KP >> 3;
  const int totChunks = NT * 16 * chunksPerRow;
  for (int idx = tid; idx < totChunks; idx += 256) {
    const int row = idx / chunksPerRow;
    const int ch  = (idx - row * chunksPerRow) << 3;
    *(v8h*)(ldsB + (long long)row * KP + ch) =
        *(const v8h*)(W + (long long)(nb + row) * KP + ch);
  }
  __syncthreads();

  const int lane = tid & 31;
  const int wave = tid >> 5;
  const int mt = blockIdx.y * 8 + wave;
  if (mt * 16 >= M) return;
  const int hs = lane >> 4;                  // half-wave select
  const int lm = lane & 15;
  const h16* arow = A + (long long)(mt * 16 + lm) * KP + (hs << 3);
  const h16* brow = ldsB + lm * KP + (hs << 3);
  v8f acc[NT];
  {
    v8f z = {};
    #pragma unroll
    for (int j = 0; j < NT; ++j) acc[j] = z;
  }
  for (int kb = 0; kb < KP; kb += 32) {
    union { v16h v; v8h h[2]; } av;
    av.h[0] = *(const v8h*)(arow + kb);        // K = kb + hs*8 + [0..7]
    av.h[1] = *(const v8h*)(arow + kb + 16);   // K = kb + hs*8 + 16 + [0..7]
    #pragma unroll
    for (int j = 0; j < NT; ++j) {
      union { v16h v; v8h h[2]; } bv;
      const h16* bp = brow + (long long)j * 16 * KP + kb;
      bv.h[0] = *(const v8h*)(bp);             // K = kb + hs*8 + [0..7]
      bv.h[1] = *(const v8h*)(bp + 16);        // K = kb + hs*8 + 16 + [0..7]
      acc[j] = __builtin_amdgcn_wmma_f32_16x16x32_f16(false, av.v, false, bv.v,
                                                      (short)0, acc[j], false, false);
    }
  }
  const int mbase = mt * 16 + (hs << 3);
  #pragma unroll
  for (int j = 0; j < NT; ++j) {
    const int ng = nb + j * 16 + lm;
    const float bi = bias ? bias[ng] : 0.f;
    const float g  = bng ? bng[ng] : 1.f;
    const float sh = bnb ? bnb[ng] : 0.f;
    for (int i = 0; i < 8; ++i) {
      float v = acc[j][i] + bi;
      if (act == 1) v = d_lrelu(v);
      v = v * g + sh;
      const long long o = (long long)(mbase + i) * NP + ng;
      if (Ch) Ch[o] = (h16)v;
      if (Cf) Cf[o] = v;
    }
  }
}

// ---- packing kernels -------------------------------------------------------
// Generic blocked weight pack: [R,C] f32 -> [RP,CP] f16 zero-padded, where
// rows/cols are remapped in (orig-block, padded-block) units (handles GRU
// gate blocks of 200->224 and concat blocks of 200->224).
__global__ void dti_packw(h16* __restrict__ dst, const float* __restrict__ src,
                          int RP, int CP, int R, int C,
                          int rbO, int rbP, int cbO, int cbP) {
  long long t = (long long)blockIdx.x * blockDim.x + threadIdx.x;
  if (t >= (long long)RP * CP) return;
  const int r = (int)(t / CP), c = (int)(t % CP);
  const int br = r / rbP, ir = r % rbP;
  const int bc = c / cbP, ic = c % cbP;
  const int orr = br * rbO + ir, occ = bc * cbO + ic;
  float v = 0.f;
  if (ir < rbO && ic < cbO && orr < R && occ < C)
    v = src[(long long)orr * C + occ];
  dst[t] = (h16)v;
}

__global__ void dti_packv(float* __restrict__ dst, const float* __restrict__ src,
                          int RP, int R, int rbO, int rbP, float scale) {
  const int r = blockIdx.x * blockDim.x + threadIdx.x;
  if (r >= RP) return;
  const int br = r / rbP, ir = r % rbP;
  const int orr = br * rbO + ir;
  dst[r] = (ir < rbO && orr < R) ? src[orr] * scale : 0.f;
}

__global__ void dti_packx(h16* __restrict__ dst, const float* __restrict__ x,
                          int N, int K, int KP) {
  long long t = (long long)blockIdx.x * blockDim.x + threadIdx.x;
  if (t >= (long long)N * KP) return;
  const int r = (int)(t / KP), k = (int)(t % KP);
  dst[t] = (k < K) ? (h16)x[(long long)r * K + k] : (h16)0.f;
}

// cat([x[src], e]) -> f16 [E,64]   (40 node feats + 10 edge feats, zero pad)
__global__ void dti_packxe(h16* __restrict__ dst, const float* __restrict__ x,
                           const float* __restrict__ ea, const int* __restrict__ src,
                           int E) {
  long long t = (long long)blockIdx.x * blockDim.x + threadIdx.x;
  if (t >= (long long)E * 64) return;
  const int e = (int)(t / 64), k = (int)(t % 64);
  float v = 0.f;
  if (k < 40)      v = x[(long long)src[e] * 40 + k];
  else if (k < 50) v = ea[(long long)e * 10 + (k - 40)];
  dst[t] = (h16)v;
}

__global__ void dti_fill(float* __restrict__ p, long long n, float v) {
  long long t = (long long)blockIdx.x * blockDim.x + threadIdx.x;
  if (t < n) p[t] = v;
}

// ---- dim-1 projections (wave-per-row dot products) -------------------------
// logits[e] = lrelu( w[0:200].A[idxA[e]] + w[200:400].Bh[idxB?idxB[e]:e] + b )
__global__ void dti_logits(const h16* __restrict__ Ah, const int* __restrict__ idxA,
                           const h16* __restrict__ Bh, const int* __restrict__ idxB,
                           const float* __restrict__ w, const float* __restrict__ bptr,
                           float* __restrict__ out, int E) {
  const int e = blockIdx.x * 8 + (threadIdx.x >> 5);
  if (e >= E) return;
  const int lane = threadIdx.x & 31;
  const long long ra = (long long)idxA[e] * GPAD;
  const long long rb = (long long)(idxB ? idxB[e] : e) * GPAD;
  float acc = 0.f;
  for (int k = lane; k < 200; k += 32)
    acc += (float)Ah[ra + k] * w[k] + (float)Bh[rb + k] * w[200 + k];
  for (int off = 16; off > 0; off >>= 1) acc += __shfl_down(acc, off, 32);
  if (lane == 0) out[e] = d_lrelu(acc + bptr[0]);
}

// out[r] = (mode==0 ? tanh : id)( w[0:Kv] . X[r] + b )
__global__ void dti_dot1(const h16* __restrict__ X, int width, int Kv,
                         const float* __restrict__ w, const float* __restrict__ bptr,
                         float* __restrict__ out, int M, int mode) {
  const int r = blockIdx.x * 8 + (threadIdx.x >> 5);
  if (r >= M) return;
  const int lane = threadIdx.x & 31;
  const long long base = (long long)r * width;
  float acc = 0.f;
  for (int k = lane; k < Kv; k += 32) acc += (float)X[base + k] * w[k];
  for (int off = 16; off > 0; off >>= 1) acc += __shfl_down(acc, off, 32);
  if (lane == 0) {
    float v = acc + bptr[0];
    out[r] = mode == 0 ? tanhf(v) : v;
  }
}

// ---- edge softmax ----------------------------------------------------------
__global__ void dti_segmax(const float* __restrict__ l, const int* __restrict__ dst,
                           float* __restrict__ m, int E) {
  const int e = blockIdx.x * blockDim.x + threadIdx.x;
  if (e < E) atomicMaxF(&m[dst[e]], l[e]);
}
__global__ void dti_expz(const float* __restrict__ l, const int* __restrict__ dst,
                         const float* __restrict__ m, float* __restrict__ z,
                         float* __restrict__ s, int E) {
  const int e = blockIdx.x * blockDim.x + threadIdx.x;
  if (e >= E) return;
  const float v = expf(l[e] - m[dst[e]]);
  z[e] = v;
  atomicAdd(&s[dst[e]], v);
}
__global__ void dti_norm(const float* __restrict__ z, const int* __restrict__ dst,
                         const float* __restrict__ s, float* __restrict__ a, int E) {
  const int e = blockIdx.x * blockDim.x + threadIdx.x;
  if (e < E) a[e] = z[e] / s[dst[e]];
}

// ctx[dst[e]] += a[e] * S[gidx ? gidx[e] : e]   (one 224-thread block per edge)
__global__ void dti_scatter(const h16* __restrict__ S, const int* __restrict__ gidx,
                            const float* __restrict__ a, const int* __restrict__ dst,
                            float* __restrict__ ctx, int E) {
  const int e = blockIdx.x, k = threadIdx.x;
  if (e >= E || k >= 200) return;
  const int row = gidx ? gidx[e] : e;
  const float v = a[e] * (float)S[(long long)row * GPAD + k];
  atomicAdd(&ctx[(long long)dst[e] * GPAD + k], v);
}

__global__ void dti_elu(const float* __restrict__ ctx, h16* __restrict__ out,
                        long long tot) {
  long long t = (long long)blockIdx.x * blockDim.x + threadIdx.x;
  if (t >= tot) return;
  const float v = ctx[t];
  out[t] = (h16)(v > 0.f ? v : expm1f(v));
}

// GRU cell + ReLU (+ optional BatchNorm), writes h (f16) and s (f32 set/accum)
__global__ void dti_gru(const float* __restrict__ gi, const float* __restrict__ gh,
                        const h16* __restrict__ hprev, h16* __restrict__ hout,
                        float* __restrict__ s, int N, int accum,
                        const float* __restrict__ bng, const float* __restrict__ bnb) {
  long long t = (long long)blockIdx.x * blockDim.x + threadIdx.x;
  if (t >= (long long)N * GPAD) return;
  const int n = (int)(t / GPAD), k = (int)(t % GPAD);
  float h = 0.f;
  if (k < 200) {
    const long long b = (long long)n * G3PAD;
    const float r  = d_sig(gi[b + k] + gh[b + k]);
    const float z  = d_sig(gi[b + 224 + k] + gh[b + 224 + k]);
    const float ng = tanhf(gi[b + 448 + k] + r * gh[b + 448 + k]);
    h = (1.f - z) * ng + z * (float)hprev[t];
    h = fmaxf(h, 0.f);
    if (bng) h = h * bng[k] * BN_SCALE + bnb[k];
  }
  hout[t] = (h16)h;
  if (accum) s[t] += h; else s[t] = h;
}

// h3[r] = concat(s1,s2)[map3[r]]  (pads already zero in s1/s2)
__global__ void dti_h3(const float* __restrict__ s1, const float* __restrict__ s2,
                       const int* __restrict__ map3, h16* __restrict__ h3, int N3) {
  long long t = (long long)blockIdx.x * blockDim.x + threadIdx.x;
  if (t >= (long long)N3 * GPAD) return;
  const int r = (int)(t / GPAD), k = (int)(t % GPAD);
  const int idx = map3[r];
  const float v = (idx < cN1) ? s1[(long long)idx * GPAD + k]
                              : s2[(long long)(idx - cN1) * GPAD + k];
  h3[t] = (h16)v;
}

// m1 input: col0 = e3, cols 1..200 = h3[src]+h3[dst], pads zero
__global__ void dti_mcat(const h16* __restrict__ h3, const float* __restrict__ e3a,
                         const int* __restrict__ src3, const int* __restrict__ dst3,
                         h16* __restrict__ out, int E) {
  long long t = (long long)blockIdx.x * blockDim.x + threadIdx.x;
  if (t >= (long long)E * GPAD) return;
  const int e = (int)(t / GPAD), k = (int)(t % GPAD);
  float v = 0.f;
  if (k == 0) v = e3a[e];
  else if (k <= 200) {
    const int c = k - 1;
    v = (float)h3[(long long)src3[e] * GPAD + c] +
        (float)h3[(long long)dst3[e] * GPAD + c];
  }
  out[t] = (h16)v;
}

// weighted-sum + max readout over graphs (one 224-thread block per edge)
__global__ void dti_readout(const h16* __restrict__ eo, const float* __restrict__ w,
                            const int* __restrict__ eg, float* __restrict__ hsum,
                            float* __restrict__ hmax, int E) {
  const int e = blockIdx.x, k = threadIdx.x;
  if (e >= E || k >= 200) return;
  const float v = (float)eo[(long long)e * GPAD + k];
  const int g = eg[e];
  atomicAdd(&hsum[(long long)g * GPAD + k], w[e] * v);
  atomicMaxF(&hmax[(long long)g * GPAD + k], v);
}

// hg = concat(hsum, hmax) packed f16 [B, 448]
__global__ void dti_hg(const float* __restrict__ hsum, const float* __restrict__ hmax,
                       h16* __restrict__ hg) {
  long long t = (long long)blockIdx.x * blockDim.x + threadIdx.x;
  if (t >= (long long)cB * G2PAD) return;
  const int b = (int)(t / G2PAD), c = (int)(t % G2PAD);
  float v = 0.f;
  if (c < 224) { if (c < 200) v = hsum[(long long)b * GPAD + c]; }
  else         { const int d = c - 224; if (d < 200) v = hmax[(long long)b * GPAD + d]; }
  hg[t] = (h16)v;
}

// ---------------------------------------------------------------------------
extern "C" void kernel_launch(void* const* d_in, const int* in_sizes, int n_in,
                              void* d_out, int out_size, void* d_ws, size_t ws_size,
                              hipStream_t stream) {
  (void)in_sizes; (void)n_in; (void)out_size; (void)ws_size;
  const float* x1  = (const float*)d_in[0];
  const float* e1  = (const float*)d_in[1];
  const int* src1  = (const int*)d_in[2];
  const int* dst1  = (const int*)d_in[3];
  const float* x2  = (const float*)d_in[4];
  const float* e2  = (const float*)d_in[5];
  const int* src2  = (const int*)d_in[6];
  const int* dst2  = (const int*)d_in[7];
  const float* e3a = (const float*)d_in[8];
  const int* src3  = (const int*)d_in[9];
  const int* dst3  = (const int*)d_in[10];
  const int* map3  = (const int*)d_in[11];
  const int* eg3   = (const int*)d_in[12];
  // d_in[13] = num_graphs (constant cB)
  const int P = 14;  // params follow in init_params() dict order
  auto prm = [&](int i) { return (const float*)d_in[P + i]; };
  const float *gc_pn_w = prm(0),  *gc_pn_b = prm(1);
  const float *gc_pe1_w = prm(2), *gc_pe1_b = prm(3);
  const float *gc_pe2_w = prm(4), *gc_pe2_b = prm(5);
  const float *gc_et_w = prm(6),  *gc_et_b = prm(7);
  const float *gc_wih = prm(8),   *gc_bih = prm(9);
  const float *gc_whh = prm(10),  *gc_bhh = prm(11);
  const float *gl_pe_w = prm(12), *gl_pe_b = prm(13);
  const float *gl_pn_w = prm(14), *gl_pn_b = prm(15);
  const float *gl_wih = prm(16),  *gl_bih = prm(17);
  const float *gl_whh = prm(18),  *gl_bhh = prm(19);
  const float *gl_bn_g = prm(20), *gl_bn_b = prm(21);
  const float *m1_w = prm(22), *m1_b = prm(23);
  const float *m2_w = prm(24), *m2_b = prm(25);
  const float *m3_w = prm(26), *m3_b = prm(27);
  const float *nc_g = prm(28), *nc_b = prm(29);
  const float *rw_w = prm(30), *rw_b = prm(31);
  const float *fc1_w = prm(32), *fc1_b = prm(33);
  const float *fc1_g = prm(34), *fc1_sb = prm(35);
  const float *fc2_w = prm(36), *fc2_b = prm(37);
  const float *fc2_g = prm(38), *fc2_sb = prm(39);
  const float *fco_w = prm(40), *fco_b = prm(41);

  // ---- workspace carve-out -------------------------------------------------
  char* wsb = (char*)d_ws;
  size_t cur = 0;
  auto alloc = [&](size_t bytes) -> void* {
    void* p = wsb + cur;
    cur += (bytes + 255) & ~(size_t)255;
    return p;
  };
  // packed weights (f16) + biases/bn (f32, padded)
  h16* pw_pn  = (h16*)alloc((size_t)GPAD * 64 * 2);   float* pb_pn  = (float*)alloc(GPAD * 4);
  h16* pw_pe1 = (h16*)alloc((size_t)GPAD * 64 * 2);   float* pb_pe1 = (float*)alloc(GPAD * 4);
  h16* pw_et  = (h16*)alloc((size_t)GPAD * GPAD * 2); float* pb_et  = (float*)alloc(GPAD * 4);
  h16* pw_wih = (h16*)alloc((size_t)G3PAD * GPAD * 2); float* pb_wih = (float*)alloc(G3PAD * 4);
  h16* pw_whh = (h16*)alloc((size_t)G3PAD * GPAD * 2); float* pb_whh = (float*)alloc(G3PAD * 4);
  h16* pw_glpn[2]; float* pb_glpn[2];
  h16* pw_glih[2]; float* pb_glih[2];
  h16* pw_glhh[2]; float* pb_glhh[2];
  for (int l = 0; l < 2; ++l) {
    pw_glpn[l] = (h16*)alloc((size_t)GPAD * GPAD * 2);  pb_glpn[l] = (float*)alloc(GPAD * 4);
    pw_glih[l] = (h16*)alloc((size_t)G3PAD * GPAD * 2); pb_glih[l] = (float*)alloc(G3PAD * 4);
    pw_glhh[l] = (h16*)alloc((size_t)G3PAD * GPAD * 2); pb_glhh[l] = (float*)alloc(G3PAD * 4);
  }
  h16* pw_m1 = (h16*)alloc((size_t)GPAD * GPAD * 2); float* pb_m1 = (float*)alloc(GPAD * 4);
  h16* pw_m2 = (h16*)alloc((size_t)GPAD * GPAD * 2); float* pb_m2 = (float*)alloc(GPAD * 4);
  h16* pw_m3 = (h16*)alloc((size_t)GPAD * GPAD * 2); float* pb_m3 = (float*)alloc(GPAD * 4);
  float* png = (float*)alloc(GPAD * 4); float* pnb = (float*)alloc(GPAD * 4);
  h16* pw_fc1 = (h16*)alloc((size_t)128 * G2PAD * 2); float* pb_fc1 = (float*)alloc(128 * 4);
  float* pg1 = (float*)alloc(128 * 4); float* ps1 = (float*)alloc(128 * 4);
  h16* pw_fc2 = (h16*)alloc((size_t)128 * 128 * 2);   float* pb_fc2 = (float*)alloc(128 * 4);
  float* pg2 = (float*)alloc(128 * 4); float* ps2 = (float*)alloc(128 * 4);
  // per-graph scratch, sized for the larger (pocket) graph
  const int Nmax = cN2, Emax = cE2;
  h16* xp   = (h16*)alloc((size_t)Nmax * 64 * 2);
  h16* xep  = (h16*)alloc((size_t)Emax * 64 * 2);
  h16* he1b = (h16*)alloc((size_t)Emax * GPAD * 2);
  h16* tbuf = (h16*)alloc((size_t)Emax * GPAD * 2);     // et / hp
  h16* hbuf = (h16*)alloc((size_t)Nmax * GPAD * 2);     // hv_new / h
  float* ctx = (float*)alloc((size_t)Nmax * GPAD * 4);
  h16* ctxh  = (h16*)alloc((size_t)Nmax * GPAD * 2);
  float* gib = (float*)alloc((size_t)Nmax * G3PAD * 4);
  float* ghb = (float*)alloc((size_t)Nmax * G3PAD * 4);
  float* lbuf = (float*)alloc((size_t)Emax * 4);
  float* zbuf = (float*)alloc((size_t)Emax * 4);
  float* abuf = (float*)alloc((size_t)Emax * 4);
  float* mseg = (float*)alloc((size_t)Nmax * 4);
  float* sseg = (float*)alloc((size_t)Nmax * 4);
  // persistent
  float* s1b = (float*)alloc((size_t)cN1 * GPAD * 4);
  float* s2b = (float*)alloc((size_t)cN2 * GPAD * 4);
  h16* h3b  = (h16*)alloc((size_t)cN3 * GPAD * 2);
  h16* minb = (h16*)alloc((size_t)cE3 * GPAD * 2);
  h16* eoA  = (h16*)alloc((size_t)cE3 * GPAD * 2);
  h16* eoB  = (h16*)alloc((size_t)cE3 * GPAD * 2);
  float* wro  = (float*)alloc((size_t)cE3 * 4);
  float* hsum = (float*)alloc((size_t)cB * GPAD * 4);
  float* hmax = (float*)alloc((size_t)cB * GPAD * 4);
  h16* hgb = (h16*)alloc((size_t)cB * G2PAD * 2);
  h16* f1b = (h16*)alloc((size_t)cB * 128 * 2);
  h16* f2b = (h16*)alloc((size_t)cB * 128 * 2);

  // ---- launch helpers ------------------------------------------------------
  auto cdiv = [](long long a, long long b) -> unsigned { return (unsigned)((a + b - 1) / b); };
  auto gemm = [&](const h16* A, const h16* Wt, const float* bias, h16* Ch, float* Cf,
                  int M, int KP, int NP, int act,
                  const float* g = nullptr, const float* b = nullptr) {
    const unsigned gy = cdiv(M, 128);
    if (NP % 112 == 0) {            // 224- and 672-wide outputs: NT=7
      dim3 grid(NP / 112, gy);
      const size_t shmem = (size_t)112 * KP * sizeof(h16);
      dti_gemm_t<7><<<grid, 256, shmem, stream>>>(A, Wt, bias, Ch, Cf, M, KP, NP, act, g, b);
    } else {                        // 128-wide FC outputs: NT=4
      dim3 grid(NP / 64, gy);
      const size_t shmem = (size_t)64 * KP * sizeof(h16);
      dti_gemm_t<4><<<grid, 256, shmem, stream>>>(A, Wt, bias, Ch, Cf, M, KP, NP, act, g, b);
    }
  };
  auto packw = [&](h16* dst, const float* src, int RP, int CP, int R, int C,
                   int rbO, int rbP, int cbO, int cbP) {
    long long tot = (long long)RP * CP;
    dti_packw<<<cdiv(tot, 256), 256, 0, stream>>>(dst, src, RP, CP, R, C, rbO, rbP, cbO, cbP);
  };
  auto packv = [&](float* dst, const float* src, int RP, int R, int rbO, int rbP,
                   float scale = 1.f) {
    dti_packv<<<cdiv(RP, 256), 256, 0, stream>>>(dst, src, RP, R, rbO, rbP, scale);
  };
  auto fill = [&](float* p, long long n, float v) {
    dti_fill<<<cdiv(n, 256), 256, 0, stream>>>(p, n, v);
  };

  // ---- pack weights (deterministic, tiny) ----------------------------------
  packw(pw_pn, gc_pn_w, GPAD, 64, 200, 40, 200, GPAD, 40, 64);     packv(pb_pn, gc_pn_b, GPAD, 200, 200, GPAD);
  packw(pw_pe1, gc_pe1_w, GPAD, 64, 200, 50, 200, GPAD, 50, 64);   packv(pb_pe1, gc_pe1_b, GPAD, 200, 200, GPAD);
  packw(pw_et, gc_et_w, GPAD, GPAD, 200, 200, 200, GPAD, 200, GPAD); packv(pb_et, gc_et_b, GPAD, 200, 200, GPAD);
  packw(pw_wih, gc_wih, G3PAD, GPAD, 600, 200, 200, GPAD, 200, GPAD); packv(pb_wih, gc_bih, G3PAD, 600, 200, GPAD);
  packw(pw_whh, gc_whh, G3PAD, GPAD, 600, 200, 200, GPAD, 200, GPAD); packv(pb_whh, gc_bhh, G3PAD, 600, 200, GPAD);
  for (int l = 0; l < 2; ++l) {
    packw(pw_glpn[l], gl_pn_w + (size_t)l * 200 * 200, GPAD, GPAD, 200, 200, 200, GPAD, 200, GPAD);
    packv(pb_glpn[l], gl_pn_b + (size_t)l * 200, GPAD, 200, 200, GPAD);
    packw(pw_glih[l], gl_wih + (size_t)l * 600 * 200, G3PAD, GPAD, 600, 200, 200, GPAD, 200, GPAD);
    packv(pb_glih[l], gl_bih + (size_t)l * 600, G3PAD, 600, 200, GPAD);
    packw(pw_glhh[l], gl_whh + (size_t)l * 600 * 200, G3PAD, GPAD, 600, 200, 200, GPAD, 200, GPAD);
    packv(pb_glhh[l], gl_bhh + (size_t)l * 600, G3PAD, 600, 200, GPAD);
  }
  packw(pw_m1, m1_w, GPAD, GPAD, 200, 201, 200, GPAD, 201, GPAD);  packv(pb_m1, m1_b, GPAD, 200, 200, GPAD);
  packw(pw_m2, m2_w, GPAD, GPAD, 200, 200, 200, GPAD, 200, GPAD);  packv(pb_m2, m2_b, GPAD, 200, 200, GPAD);
  packw(pw_m3, m3_w, GPAD, GPAD, 200, 200, 200, GPAD, 200, GPAD);  packv(pb_m3, m3_b, GPAD, 200, 200, GPAD);
  packv(png, nc_g, GPAD, 200, 200, GPAD, BN_SCALE);                packv(pnb, nc_b, GPAD, 200, 200, GPAD);
  packw(pw_fc1, fc1_w, 128, G2PAD, 128, 400, 128, 128, 200, GPAD); packv(pb_fc1, fc1_b, 128, 128, 128, 128);
  packv(pg1, fc1_g, 128, 128, 128, 128, BN_SCALE);                 packv(ps1, fc1_sb, 128, 128, 128, 128);
  packw(pw_fc2, fc2_w, 128, 128, 128, 128, 128, 128, 128, 128);    packv(pb_fc2, fc2_b, 128, 128, 128, 128);
  packv(pg2, fc2_g, 128, 128, 128, 128, BN_SCALE);                 packv(ps2, fc2_sb, 128, 128, 128, 128);

  // ---- edge softmax helper -------------------------------------------------
  auto softmax = [&](const float* logit, const int* dst, int N, int E) {
    fill(mseg, N, -3.4e38f);
    fill(sseg, N, 0.f);
    dti_segmax<<<cdiv(E, 256), 256, 0, stream>>>(logit, dst, mseg, E);
    dti_expz<<<cdiv(E, 256), 256, 0, stream>>>(logit, dst, mseg, zbuf, sseg, E);
    dti_norm<<<cdiv(E, 256), 256, 0, stream>>>(zbuf, dst, sseg, abuf, E);
  };

  // ---- one covalent-graph tower (GetContext + 2 GNN layers) ---------------
  auto cov = [&](const float* x, const float* ea, const int* src, const int* dst,
                 int N, int E, float* sOut) {
    dti_packx<<<cdiv((long long)N * 64, 256), 256, 0, stream>>>(xp, x, N, 40, 64);
    gemm(xp, pw_pn, pb_pn, hbuf, nullptr, N, 64, GPAD, 1);                 // hv_new
    dti_packxe<<<cdiv((long long)E * 64, 256), 256, 0, stream>>>(xep, x, ea, src, E);
    gemm(xep, pw_pe1, pb_pe1, he1b, nullptr, E, 64, GPAD, 1);              // he1
    dti_logits<<<cdiv(E, 8), 256, 0, stream>>>(hbuf, dst, he1b, nullptr,
                                               gc_pe2_w, gc_pe2_b, lbuf, E);
    softmax(lbuf, dst, N, E);
    gemm(he1b, pw_et, pb_et, tbuf, nullptr, E, GPAD, GPAD, 0);             // et(he1)
    fill(ctx, (long long)N * GPAD, 0.f);
    dti_scatter<<<E, 224, 0, stream>>>(tbuf, nullptr, abuf, dst, ctx, E);
    dti_elu<<<cdiv((long long)N * GPAD, 256), 256, 0, stream>>>(ctx, ctxh, (long long)N * GPAD);
    gemm(ctxh, pw_wih, pb_wih, nullptr, gib, N, GPAD, G3PAD, 0);
    gemm(hbuf, pw_whh, pb_whh, nullptr, ghb, N, GPAD, G3PAD, 0);
    dti_gru<<<cdiv((long long)N * GPAD, 256), 256, 0, stream>>>(
        gib, ghb, hbuf, hbuf, sOut, N, 0, nullptr, nullptr);               // h, s=h
    for (int l = 0; l < 2; ++l) {
      dti_logits<<<cdiv(E, 8), 256, 0, stream>>>(hbuf, dst, hbuf, src,
                                                 gl_pe_w + (size_t)l * 400,
                                                 gl_pe_b + l, lbuf, E);
      softmax(lbuf, dst, N, E);
      gemm(hbuf, pw_glpn[l], pb_glpn[l], tbuf, nullptr, N, GPAD, GPAD, 0); // hp
      fill(ctx, (long long)N * GPAD, 0.f);
      dti_scatter<<<E, 224, 0, stream>>>(tbuf, src, abuf, dst, ctx, E);
      dti_elu<<<cdiv((long long)N * GPAD, 256), 256, 0, stream>>>(ctx, ctxh, (long long)N * GPAD);
      gemm(ctxh, pw_glih[l], pb_glih[l], nullptr, gib, N, GPAD, G3PAD, 0);
      gemm(hbuf, pw_glhh[l], pb_glhh[l], nullptr, ghb, N, GPAD, G3PAD, 0);
      dti_gru<<<cdiv((long long)N * GPAD, 256), 256, 0, stream>>>(
          gib, ghb, hbuf, hbuf, sOut, N, 1,
          gl_bn_g + (size_t)l * 200, gl_bn_b + (size_t)l * 200);           // h, s+=bn(h)
    }
  };

  cov(x1, e1, src1, dst1, cN1, cE1, s1b);
  cov(x2, e2, src2, dst2, cN2, cE2, s2b);

  // ---- interaction graph ---------------------------------------------------
  dti_h3<<<cdiv((long long)cN3 * GPAD, 256), 256, 0, stream>>>(s1b, s2b, map3, h3b, cN3);
  dti_mcat<<<cdiv((long long)cE3 * GPAD, 256), 256, 0, stream>>>(h3b, e3a, src3, dst3, minb, cE3);
  gemm(minb, pw_m1, pb_m1, eoA, nullptr, cE3, GPAD, GPAD, 1);
  gemm(eoA,  pw_m2, pb_m2, eoB, nullptr, cE3, GPAD, GPAD, 1);
  gemm(eoB,  pw_m3, pb_m3, eoA, nullptr, cE3, GPAD, GPAD, 1, png, pnb);    // + nc BN

  // ---- readout + FC head ---------------------------------------------------
  dti_dot1<<<cdiv(cE3, 8), 256, 0, stream>>>(eoA, GPAD, 200, rw_w, rw_b, wro, cE3, 0);
  fill(hsum, (long long)cB * GPAD, 0.f);
  fill(hmax, (long long)cB * GPAD, -3.4e38f);
  dti_readout<<<cE3, 224, 0, stream>>>(eoA, wro, eg3, hsum, hmax, cE3);
  dti_hg<<<cdiv((long long)cB * G2PAD, 256), 256, 0, stream>>>(hsum, hmax, hgb);
  gemm(hgb, pw_fc1, pb_fc1, f1b, nullptr, cB, G2PAD, 128, 1, pg1, ps1);
  gemm(f1b, pw_fc2, pb_fc2, f2b, nullptr, cB, 128, 128, 1, pg2, ps2);
  dti_dot1<<<cdiv(cB, 8), 256, 0, stream>>>(f2b, 128, 128, fco_w, fco_b,
                                            (float*)d_out, cB, 1);
}
